// LambdaConv_20718922236208
// MI455X (gfx1250) — matrix operand order repaired and checked by reading, not compiled, not run on hardware
//
#include <hip/hip_runtime.h>
#include <hip/hip_bf16.h>

// ---------------------------------------------------------------------------
// LambdaConv for MI455X (gfx1250, wave32, WMMA).
// Hot path: 23x23 positional conv (35.5 GFLOP) as im2col GEMM on
// v_wmma_f32_16x16x32_f16. emb A-fragment reused across 4 N-tiles per conv
// row; all fragment elements fetched as aligned 32-bit pair loads (dual LDS
// image, one shifted by a halfword, so every lane has an even base);
// v-slice staged by 16B-aligned global_load_async_to_lds_b128; fused with
// the y_p/y_c output contraction so lambda_p is never materialized.
// ---------------------------------------------------------------------------

typedef __attribute__((ext_vector_type(16))) _Float16 v16h;
typedef __attribute__((ext_vector_type(8)))  float    v8f;

union Frag { v16h h; unsigned u[8]; };

#define EPS     1e-5f
#define BATCH   8
#define CIN     256
#define WH      64
#define NPOS    4096      // 64*64
#define HEADS   4
#define KK      16
#define VV      64
#define MM      23
#define PAD     11
#define ROWP    32        // conv row padded 23 -> 32 taps
#define TAPS_P  (MM*ROWP) // 736
#define CH_ALL  144       // 64 q + 16 k + 64 v stacked channel rows
// padded LDS image: 86 rows x 104 cols; data starts at col 16 (16B-aligned
// async stores); col for tap j = pj + dj - 11  ->  element col pj + dj + 5.
#define VROWS   86
#define VCOLS   104
#define VOFFS   16        // data start column
#define VIMG    (VROWS*VCOLS)

// ---- WMMA fragment index maps (contiguous k = 16h+e used by this file; A and B match, wave32) --------------
// A (16x32 f16): lane<16: row M=lane, K=e; lane>=16: row M=lane-16, K=16+e
__device__ __forceinline__ int a_kmap(int lane, int e) {
  return (lane < 16) ? e : e + 16;
}
// B (32x16 f16): lane<16: col N=lane, K=e; lane>=16: col N=lane-16, K=16+e
__device__ __forceinline__ int b_kmap(int lane, int e) {
  return (lane < 16) ? e : e + 16;
}
// C/D (16x16 f32): elem r: M = (lane<16)? r : r+8 ; N = lane&15

// ---------------------------------------------------------------------------
// Kernel 1a: embedding (16 x 23 x 23 f32) -> f16, each 23-tap row padded to 32
// ---------------------------------------------------------------------------
__global__ void emb_to_f16(const float* __restrict__ emb, _Float16* __restrict__ embh) {
  for (int i = threadIdx.x; i < KK * TAPS_P; i += blockDim.x) {
    int k  = i / TAPS_P, r = i % TAPS_P;
    int di = r / ROWP,  dj = r % ROWP;
    embh[i] = (dj < MM) ? (_Float16)emb[(k * MM + di) * MM + dj] : (_Float16)0.0f;
  }
}

// ---------------------------------------------------------------------------
// Kernel 1b: pack [Wq;Wk;Wv] -> f16, ch-major [ch][k] (paired-k loads)
// ---------------------------------------------------------------------------
__global__ void pack_wt(const float* __restrict__ Wq, const float* __restrict__ Wk,
                        const float* __restrict__ Wv, _Float16* __restrict__ Wt) {
  int i = blockIdx.x * blockDim.x + threadIdx.x;       // over 144*256
  if (i >= CIN * CH_ALL) return;
  int ch = i / CIN, k = i % CIN;
  float w;
  if (ch < 64)      w = Wq[ch * CIN + k];
  else if (ch < 80) w = Wk[(ch - 64) * CIN + k];
  else              w = Wv[(ch - 80) * CIN + k];
  Wt[ch * CIN + k] = (_Float16)w;
}

// ---------------------------------------------------------------------------
// Kernel 2: projection GEMM  [Wq;Wk;Wv](144x256) x x_b(256x4096), fused BN.
// Block = 288 threads = 9 waves = all 9 M-tiles of one (b, n-tile).
// x chunk staged n-major in LDS as f16; A/B fragments via paired b32 loads.
// q written position-major: qbuf[b][n][64] (vectorized conv epilogue reads).
// ---------------------------------------------------------------------------
__global__ void __launch_bounds__(288)
proj_gemm(const float* __restrict__ x, const _Float16* __restrict__ Wt,
          const float* __restrict__ qg, const float* __restrict__ qb_,
          const float* __restrict__ qm, const float* __restrict__ qv,
          const float* __restrict__ vg, const float* __restrict__ vb_,
          const float* __restrict__ vm, const float* __restrict__ vv_,
          float* __restrict__ qbuf, float* __restrict__ kbuf,
          _Float16* __restrict__ vh) {
  __shared__ __align__(16) _Float16 s_x[16 * 32];       // n-major: s_x[n][k]

  const int tid  = threadIdx.x;
  const int lane = tid & 31;
  const int wv   = tid >> 5;                            // 0..8 -> m-tile
  const int b    = blockIdx.x >> 8;                     // 2048 = 8 b * 256 n
  const int n0   = (blockIdx.x & 255) * 16;
  const int m0   = wv * 16;

  const int lane15 = lane & 15;
  const float* xb  = x + (size_t)b * CIN * NPOS;
  const _Float16* wrow = Wt + (size_t)(m0 + lane15) * CIN;

  v8f c = {};
  for (int k0 = 0; k0 < CIN; k0 += 32) {
    __syncthreads();
    for (int idx = tid; idx < 512; idx += 288) {        // stage + cvt to f16
      int kk = idx >> 4, nn = idx & 15;                 // global reads n-fast
      s_x[nn * 32 + kk] = (_Float16)xb[(size_t)(k0 + kk) * NPOS + n0 + nn];
    }
    __syncthreads();

    Frag fa, fb;
#pragma unroll
    for (int e2 = 0; e2 < 8; ++e2) {
      fa.u[e2] = *(const unsigned*)&wrow[k0 + a_kmap(lane, 2 * e2)];
      fb.u[e2] = *(const unsigned*)&s_x[lane15 * 32 + b_kmap(lane, 2 * e2)];
    }
    c = __builtin_amdgcn_wmma_f32_16x16x32_f16(false, fa.h, false, fb.h,
                                               (short)0, c, false, false);
  }

  const int bn = n0 + lane15;
#pragma unroll
  for (int r = 0; r < 8; ++r) {
    int m  = (lane < 16) ? r : r + 8;
    int ch = m0 + m;
    float val = c[r];
    if (ch < 64) {
      float sc = qg[ch] * rsqrtf(qv[ch] + EPS);
      val = (val - qm[ch]) * sc + qb_[ch];
      qbuf[((size_t)b * NPOS + bn) * 64 + ch] = val;    // position-major
    } else if (ch < 80) {
      kbuf[((size_t)b * KK + (ch - 64)) * NPOS + bn] = val;
    } else {
      int c2 = ch - 80;
      float sc = vg[c2] * rsqrtf(vv_[c2] + EPS);
      val = (val - vm[c2]) * sc + vb_[c2];
      vh[((size_t)b * VV + c2) * NPOS + bn] = (_Float16)val;
    }
  }
}

// ---------------------------------------------------------------------------
// Kernel 3: softmax over n=4096 per (b,k) row, in place. 128 rows.
// ---------------------------------------------------------------------------
__global__ void softmax_rows(float* __restrict__ kf) {
  __shared__ float red[256];
  const int tid = threadIdx.x;
  float* row = kf + (size_t)blockIdx.x * NPOS;

  float mx = -3.4e38f;
  for (int i = tid; i < NPOS; i += 256) mx = fmaxf(mx, row[i]);
  red[tid] = mx; __syncthreads();
  for (int s = 128; s > 0; s >>= 1) {
    if (tid < s) red[tid] = fmaxf(red[tid], red[tid + s]);
    __syncthreads();
  }
  mx = red[0]; __syncthreads();

  float sum = 0.f;
  for (int i = tid; i < NPOS; i += 256) sum += __expf(row[i] - mx);
  red[tid] = sum; __syncthreads();
  for (int s = 128; s > 0; s >>= 1) {
    if (tid < s) red[tid] += red[tid + s];
    __syncthreads();
  }
  float inv = 1.0f / red[0];
  for (int i = tid; i < NPOS; i += 256) row[i] = __expf(row[i] - mx) * inv;
}

// ---------------------------------------------------------------------------
// Kernel 4: lambda_c[b] (16x64) = kf (16x4096) x v^T via WMMA, K=4096.
// ---------------------------------------------------------------------------
__global__ void lambda_c_gemm(const float* __restrict__ kf,
                              const _Float16* __restrict__ vh,
                              float* __restrict__ lamc) {
  const int lane = threadIdx.x & 31;
  const int g    = blockIdx.x * 4 + (threadIdx.x >> 5);
  const int b    = g >> 2;
  const int v0   = (g & 3) * 16;

  const float*    kfb = kf + (size_t)b * KK * NPOS;
  const _Float16* vhb = vh + (size_t)b * VV * NPOS;
  const int am = lane & 15;
  const int bv = v0 + (lane & 15);

  v8f c = {};
  for (int k0 = 0; k0 < NPOS; k0 += 32) {
    v16h a;
    Frag fb;
#pragma unroll
    for (int e = 0; e < 16; ++e)
      a[e] = (_Float16)kfb[(size_t)am * NPOS + k0 + a_kmap(lane, e)];
#pragma unroll
    for (int e2 = 0; e2 < 8; ++e2)
      fb.u[e2] = *(const unsigned*)&vhb[(size_t)bv * NPOS + k0 + b_kmap(lane, 2 * e2)];
    c = __builtin_amdgcn_wmma_f32_16x16x32_f16(false, a, false, fb.h,
                                               (short)0, c, false, false);
  }
#pragma unroll
  for (int r = 0; r < 8; ++r) {
    int m = (lane < 16) ? r : r + 8;
    lamc[((size_t)b * KK + m) * VV + v0 + (lane & 15)] = c[r];
  }
}

// ---------------------------------------------------------------------------
// Kernel 5: fused positional conv + output. Block (256 thr, 8 waves) owns one
// (b, vv). v-slice async-copied (16B-aligned b128) into a zero-halo LDS image;
// a halfword-shifted twin image makes every lane's paired b32 B-load aligned.
// Wave processes a full 64-wide image row: emb A-fragment loaded once per
// conv row (8 paired b32) and reused by 4 WMMAs. Epilogue contracts with q
// (float4 loads, position-major) and lambda_c; half-wave k-sums combined via
// wave32 shfl_xor(16).
// ---------------------------------------------------------------------------
__global__ void conv_fused_out(const _Float16* __restrict__ embh,
                               const _Float16* __restrict__ vh,
                               const float* __restrict__ qbuf,
                               const float* __restrict__ lamc,
                               float* __restrict__ out) {
  __shared__ __align__(16) _Float16 s_emb[KK * TAPS_P]; // 23552 B
  __shared__ __align__(16) _Float16 s_v[VIMG];          // 17888 B
  __shared__ __align__(16) _Float16 s_v2[VIMG];         // 17888 B (shift +1)

  const int tid  = threadIdx.x;
  const int lane = tid & 31;
  const int wv   = tid >> 5;
  const int b    = blockIdx.x >> 6;         // 512 blocks = 8 b * 64 vv
  const int vvch = blockIdx.x & 63;

  const _Float16* vslice = vh + ((size_t)b * VV + vvch) * NPOS;

  // stage emb + zero the halo image
  for (int i = tid; i < KK * TAPS_P; i += 256) s_emb[i] = embh[i];
  for (int i = tid; i < VIMG; i += 256) s_v[i] = (_Float16)0.0f;
  __syncthreads();

  // async-copy v rows into the LDS interior (cols 16..79): 16B-aligned b128
  for (int cidx = tid; cidx < 512; cidx += 256) {
    int row = cidx >> 3, ch = cidx & 7;
    unsigned lds_off = (unsigned)(size_t)(&s_v[(row + PAD) * VCOLS + VOFFS + ch * 8]);
    const _Float16* gsrc = vslice + row * WH + ch * 8;
    asm volatile("global_load_async_to_lds_b128 %0, %1, off"
                 :: "v"(lds_off), "v"(gsrc) : "memory");
  }
  asm volatile("s_wait_asynccnt 0" ::: "memory");
  __syncthreads();

  // build halfword-shifted twin: s_v2[i] = s_v[i+1]
  for (int i = tid; i < VIMG; i += 256)
    s_v2[i] = (i + 1 < VIMG) ? s_v[i + 1] : (_Float16)0.0f;
  __syncthreads();

  const int lane15 = lane & 15;
  const int am     = lane15;                // emb row (k) this lane holds in A
  const int bofs_h = (lane < 16) ? 0 : 16;  // K offset for hi half-wave
  // element col for tap dj at position col pj:  pj + dj + (VOFFS - PAD) = +5
  const int cbase  = 5 + lane15 + bofs_h;
  const int par    = cbase & 1;             // odd base -> use shifted image
  const _Float16* vb = par ? s_v2 : s_v;
  const int cb     = cbase - par;

  const float* lc = lamc + (size_t)b * KK * VV;
  const size_t ob = (size_t)b * (HEADS * VV) * NPOS;

  float lcreg[8];
#pragma unroll
  for (int r = 0; r < 8; ++r) {
    int m = (lane < 16) ? r : r + 8;
    lcreg[r] = lc[m * VV + vvch];
  }

  for (int grp = 0; grp < 8; ++grp) {
    const int prow = wv * 8 + grp;          // image row 0..63
    const int n0   = prow * WH;

    v8f c[4] = {};
    for (int di = 0; di < MM; ++di) {
      Frag fa, f0, f1, f2, f3;
      const int abase = am * TAPS_P + di * ROWP;
      const int rbase = (prow + di) * VCOLS + cb;
#pragma unroll
      for (int e2 = 0; e2 < 8; ++e2) {
        const int cc = 2 * e2;
        fa.u[e2] = *(const unsigned*)&s_emb[abase + a_kmap(lane, cc)];
        f0.u[e2] = *(const unsigned*)&vb[rbase + cc];
        f1.u[e2] = *(const unsigned*)&vb[rbase + 16 + cc];
        f2.u[e2] = *(const unsigned*)&vb[rbase + 32 + cc];
        f3.u[e2] = *(const unsigned*)&vb[rbase + 48 + cc];
      }
      c[0] = __builtin_amdgcn_wmma_f32_16x16x32_f16(false, fa.h, false, f0.h,
                                                    (short)0, c[0], false, false);
      c[1] = __builtin_amdgcn_wmma_f32_16x16x32_f16(false, fa.h, false, f1.h,
                                                    (short)0, c[1], false, false);
      c[2] = __builtin_amdgcn_wmma_f32_16x16x32_f16(false, fa.h, false, f2.h,
                                                    (short)0, c[2], false, false);
      c[3] = __builtin_amdgcn_wmma_f32_16x16x32_f16(false, fa.h, false, f3.h,
                                                    (short)0, c[3], false, false);
    }

#pragma unroll
    for (int t = 0; t < 4; ++t) {
      const int p = n0 + 16 * t + lane15;
      float lam[8];
#pragma unroll
      for (int r = 0; r < 8; ++r) lam[r] = c[t][r] + lcreg[r];

      // q is position-major: 64 consecutive floats per position; this lane's
      // half-wave reads ch = h*16 + {0..7 | 8..15} as two float4s per head.
      const float* qrow = qbuf + ((size_t)b * NPOS + p) * 64 + ((lane < 16) ? 0 : 8);
      float acc[4];
#pragma unroll
      for (int h = 0; h < 4; ++h) {
        const float4 qa = *(const float4*)(qrow + h * 16);
        const float4 qb4 = *(const float4*)(qrow + h * 16 + 4);
        acc[h] = qa.x * lam[0] + qa.y * lam[1] + qa.z * lam[2] + qa.w * lam[3]
               + qb4.x * lam[4] + qb4.y * lam[5] + qb4.z * lam[6] + qb4.w * lam[7];
      }
#pragma unroll
      for (int h = 0; h < 4; ++h) acc[h] += __shfl_xor(acc[h], 16, 32);
      if (lane < 16) {
#pragma unroll
        for (int h = 0; h < 4; ++h)
          out[ob + (size_t)(h * VV + vvch) * NPOS + p] = acc[h];
      }
    }
  }
}

// ---------------------------------------------------------------------------
extern "C" void kernel_launch(void* const* d_in, const int* in_sizes, int n_in,
                              void* d_out, int out_size, void* d_ws, size_t ws_size,
                              hipStream_t stream) {
  const float* x   = (const float*)d_in[0];
  const float* Wq  = (const float*)d_in[1];
  const float* qg  = (const float*)d_in[2];
  const float* qb_ = (const float*)d_in[3];
  const float* qm  = (const float*)d_in[4];
  const float* qv  = (const float*)d_in[5];
  const float* Wk  = (const float*)d_in[6];
  const float* Wv  = (const float*)d_in[7];
  const float* vg  = (const float*)d_in[8];
  const float* vb_ = (const float*)d_in[9];
  const float* vm  = (const float*)d_in[10];
  const float* vv_ = (const float*)d_in[11];
  const float* emb = (const float*)d_in[12];
  float* out = (float*)d_out;

  // workspace layout (bytes, 256-aligned)
  char* w = (char*)d_ws;
  float*    qbuf = (float*)(w + 0);               //  8 MB: q (8,4096,64) f32
  float*    kbuf = (float*)(w + 8388608);         //  2 MB: k logits -> kf
  _Float16* vh   = (_Float16*)(w + 10485760);     //  4 MB: v (8,64,4096) f16
  _Float16* embh = (_Float16*)(w + 14680064);     // 23 KB: emb f16 row-padded
  float*    lamc = (float*)(w + 14703616);        // 32 KB: lambda_c (8,16,64)
  _Float16* Wt   = (_Float16*)(w + 14736384);     // 72 KB: packed f16 weights

  emb_to_f16<<<1, 256, 0, stream>>>(emb, embh);
  pack_wt<<<(CIN * CH_ALL + 255) / 256, 256, 0, stream>>>(Wq, Wk, Wv, Wt);

  proj_gemm<<<2048, 288, 0, stream>>>(x, Wt, qg, qb_, qm, qv,
                                      vg, vb_, vm, vv_, qbuf, kbuf, vh);

  softmax_rows<<<BATCH * KK, 256, 0, stream>>>(kbuf);

  lambda_c_gemm<<<8, 128, 0, stream>>>(kbuf, vh, lamc);

  conv_fused_out<<<512, 256, 0, stream>>>(embh, vh, qbuf, lamc, out);
}
